// TemporalContrastiveLoss_10780367913244
// MI455X (gfx1250) — matrix operand order repaired and checked by reading, not compiled, not run on hardware
//
#include <hip/hip_runtime.h>
#include <hip/hip_bf16.h>

// ---------------------------------------------------------------------------
// TemporalContrastiveLoss on MI455X (gfx1250, wave32, WMMA)
//   N = M = 8192, H = 1024, T = 0.07
//   out[0] = mean_i log( sum_j exp((sim_ij - max_i)/T) )      (contrastive)
//   out[1] = 1 - mean_i dot(That[pos_i], That[pos_{i+1}])     (contextual)
// GEMM fused with online softmax/argmax -> never materializes 8192x8192 sim.
// f16 operands (both fit in 192MB L2: 2x16MB), f32 accumulation via
// v_wmma_f32_16x16x32_f16. Columns split into 4 chunks per row-strip
// (2048 waves) with a partial-state merge pass for occupancy.
// ---------------------------------------------------------------------------

typedef __attribute__((ext_vector_type(16))) _Float16 v16h;
typedef __attribute__((ext_vector_type(8)))  _Float16 v8h;
typedef __attribute__((ext_vector_type(4)))  _Float16 v4h;
typedef __attribute__((ext_vector_type(8)))  float    v8f;

#define NROWS  8192
#define NCOLS  8192
#define HDIM   1024
#define NCHUNK 4
#define CHUNKC (NCOLS / NCHUNK)            // 2048 columns per chunk
#define INV_T  14.285714285714286f         // 1/0.07

__device__ __forceinline__ v8f wmma_f16(v16h a, v16h b, v8f c) {
  // (neg_a, A, neg_b, B, c_mod, C, reuse_a, reuse_b)
  return __builtin_amdgcn_wmma_f32_16x16x32_f16(false, a, false, b,
                                                (short)0, c, false, false);
}

// --------------------------- init / finalize -------------------------------

__global__ void init_kernel(float* __restrict__ acc) {
  if (threadIdx.x == 0) { acc[0] = 0.0f; acc[1] = 0.0f; }
}

__global__ void fin_kernel(const float* __restrict__ acc, float* __restrict__ out) {
  if (threadIdx.x == 0) {
    out[0] = acc[0] * (1.0f / (float)NROWS);           // mean log-sum-exp
    out[1] = 1.0f - acc[1] * (1.0f / (float)(NROWS - 1));
  }
}

// --------------------------- row L2-normalize ------------------------------
// one 256-thread block per row; 4 floats per thread (H = 1024)

__global__ __launch_bounds__(256) void nrm_kernel(const float* __restrict__ in,
                                                  _Float16* __restrict__ out) {
  const int row = blockIdx.x;
  const int t = threadIdx.x;
  const float4 v = ((const float4*)(in + (size_t)row * HDIM))[t];
  float ss = v.x * v.x + v.y * v.y + v.z * v.z + v.w * v.w;
#pragma unroll
  for (int d = 16; d >= 1; d >>= 1) ss += __shfl_xor(ss, d, 32);
  __shared__ float red[8];
  if ((t & 31) == 0) red[t >> 5] = ss;
  __syncthreads();
  float tot = red[0];
#pragma unroll
  for (int i = 1; i < 8; ++i) tot += red[i];
  const float sc = 1.0f / fmaxf(sqrtf(tot), 1e-12f);
  v4h h;
  h[0] = (_Float16)(v.x * sc);
  h[1] = (_Float16)(v.y * sc);
  h[2] = (_Float16)(v.z * sc);
  h[3] = (_Float16)(v.w * sc);
  ((v4h*)(out + (size_t)row * HDIM))[t] = h;
}

// ------------------ fused GEMM + online softmax/argmax ---------------------
// 256 blocks x 256 threads = 2048 waves. Each wave owns a 16-row strip and a
// 2048-column chunk; per column block of 64: 4 WMMA tiles share each A fetch.
// Partial (max, expsum, argmax) per row/chunk written for the merge pass.

__global__ __launch_bounds__(256) void simred_kernel(
    const _Float16* __restrict__ As, const _Float16* __restrict__ Bt,
    float* __restrict__ partM, float* __restrict__ partS,
    int* __restrict__ partI) {
  const int lane    = threadIdx.x & 31;
  const int l15     = lane & 15;
  const int halfSel = lane >> 4;                 // 0: rows 0-7, 1: rows 8-15
  const int wgid    = blockIdx.x * 8 + (threadIdx.x >> 5);   // 0..2047
  const int rowBase = (wgid >> 2) * 16;
  const int chunk   = wgid & (NCHUNK - 1);
  const int colStart = chunk * CHUNKC;

  // A layout (16-bit A 16x32): lane row = l15; halves 0-15 at K = kb+halfSel*16..+15
  // (matches B).
  const _Float16* aRow = As + (size_t)(rowBase + l15) * HDIM + halfSel * 16;

  // per-lane online stats: rows (halfSel*8 + r), columns c with c%16 == l15
  float m[8], s[8];
  int   ix[8];
#pragma unroll
  for (int r = 0; r < 8; ++r) { m[r] = -3.4e38f; s[r] = 0.0f; ix[r] = colStart; }

  for (int cb = colStart; cb < colStart + CHUNKC; cb += 64) {
    v8f c0 = {}, c1 = {}, c2 = {}, c3 = {};
    // B layout (16-bit B 32x16): lane col = l15; halves = K kb+halfSel*16..+15
    const _Float16* bBase = Bt + (size_t)(cb + l15) * HDIM + halfSel * 16;

#pragma unroll 8
    for (int kb = 0; kb < HDIM; kb += 32) {
      const v8h alo = *(const v8h*)(aRow + kb);
      const v8h ahi = *(const v8h*)(aRow + kb + 8);
      const v16h a = __builtin_shufflevector(alo, ahi, 0, 1, 2, 3, 4, 5, 6, 7,
                                             8, 9, 10, 11, 12, 13, 14, 15);
      c0 = wmma_f16(a, *(const v16h*)(bBase + 0 * 16 * HDIM + kb), c0);
      c1 = wmma_f16(a, *(const v16h*)(bBase + 1 * 16 * HDIM + kb), c1);
      c2 = wmma_f16(a, *(const v16h*)(bBase + 2 * 16 * HDIM + kb), c2);
      c3 = wmma_f16(a, *(const v16h*)(bBase + 3 * 16 * HDIM + kb), c3);
    }

    // branchless online max/argmax/exp-sum update, 32 sims per lane
#pragma unroll
    for (int t = 0; t < 4; ++t) {
      const v8f cc = (t == 0) ? c0 : (t == 1) ? c1 : (t == 2) ? c2 : c3;
      const int col = cb + t * 16 + l15;
#pragma unroll
      for (int r = 0; r < 8; ++r) {
        const float x = cc[r];
        const float e = __expf(-fabsf(x - m[r]) * INV_T); // exp(-|x-m|/T)
        const bool  g = x > m[r];
        s[r]  = g ? fmaf(s[r], e, 1.0f) : (s[r] + e);
        ix[r] = g ? col : ix[r];
        m[r]  = g ? x : m[r];
      }
    }
  }

  // merge across the 16 lanes of each half (xor<16 never crosses bit 4)
#pragma unroll
  for (int d = 1; d < 16; d <<= 1) {
#pragma unroll
    for (int r = 0; r < 8; ++r) {
      const float om = __shfl_xor(m[r], d, 32);
      const float os = __shfl_xor(s[r], d, 32);
      const int   oi = __shfl_xor(ix[r], d, 32);
      const float M  = fmaxf(m[r], om);
      const float ns = s[r] * __expf((m[r] - M) * INV_T) +
                       os   * __expf((om   - M) * INV_T);
      const bool take = (om > m[r]) || (om == m[r] && oi < ix[r]); // first-max
      ix[r] = take ? oi : ix[r];
      m[r] = M;
      s[r] = ns;
    }
  }

  if (l15 == 0) {
    const int rbase = rowBase + halfSel * 8;
#pragma unroll
    for (int r = 0; r < 8; ++r) {
      const int o = (rbase + r) * NCHUNK + chunk;
      partM[o] = m[r];
      partS[o] = s[r];
      partI[o] = ix[r];
    }
  }
}

// -------------- merge per-row partial softmax/argmax states ----------------
// chunks are in ascending column order -> strict '>' keeps first-max index

__global__ __launch_bounds__(256) void merge_kernel(
    const float* __restrict__ partM, const float* __restrict__ partS,
    const int* __restrict__ partI, float* __restrict__ logS,
    int* __restrict__ pos) {
  const int row = blockIdx.x * 256 + threadIdx.x;
  float m = partM[row * NCHUNK];
  float s = partS[row * NCHUNK];
  int  ix = partI[row * NCHUNK];
#pragma unroll
  for (int c = 1; c < NCHUNK; ++c) {
    const float mc = partM[row * NCHUNK + c];
    const float sc = partS[row * NCHUNK + c];
    const int   ic = partI[row * NCHUNK + c];
    const float M  = fmaxf(m, mc);
    s  = s * __expf((m - M) * INV_T) + sc * __expf((mc - M) * INV_T);
    ix = (mc > m) ? ic : ix;
    m  = M;
  }
  logS[row] = __logf(s);   // -logp at argmax == log(sum exp((x-max)/T))
  pos[row]  = ix;
}

// ---------------- contextual consistency: adjacent NN dots -----------------
// one wave per (i, i+1) pair; f32-accumulated dot of two f16 rows

__global__ __launch_bounds__(256) void ctx_kernel(const _Float16* __restrict__ Bt,
                                                  const int* __restrict__ pos,
                                                  float* __restrict__ accNN) {
  const int w    = (int)((blockIdx.x * blockDim.x + threadIdx.x) >> 5);
  const int lane = threadIdx.x & 31;
  if (w >= NROWS - 1) return;
  const _Float16* p = Bt + (size_t)pos[w]     * HDIM;
  const _Float16* q = Bt + (size_t)pos[w + 1] * HDIM;
  float acc = 0.0f;
#pragma unroll
  for (int k = lane * 8; k < HDIM; k += 32 * 8) {
    const v8h a = *(const v8h*)(p + k);
    const v8h b = *(const v8h*)(q + k);
#pragma unroll
    for (int i = 0; i < 8; ++i) acc += (float)a[i] * (float)b[i];
  }
#pragma unroll
  for (int d = 16; d >= 1; d >>= 1) acc += __shfl_xor(acc, d, 32);
  if (lane == 0) atomicAdd(accNN, acc);
}

// ----------------------- logS reduction (8192 values) ----------------------

__global__ __launch_bounds__(256) void logsum_kernel(const float* __restrict__ logS,
                                                     float* __restrict__ accLog) {
  const int i = blockIdx.x * 256 + threadIdx.x;
  float v = logS[i];
#pragma unroll
  for (int d = 16; d >= 1; d >>= 1) v += __shfl_xor(v, d, 32);
  __shared__ float red[8];
  if ((threadIdx.x & 31) == 0) red[threadIdx.x >> 5] = v;
  __syncthreads();
  if (threadIdx.x == 0) {
    float tot = red[0];
#pragma unroll
    for (int j = 1; j < 8; ++j) tot += red[j];
    atomicAdd(accLog, tot);
  }
}

// ------------------------------- launcher ----------------------------------

extern "C" void kernel_launch(void* const* d_in, const int* in_sizes, int n_in,
                              void* d_out, int out_size, void* d_ws, size_t ws_size,
                              hipStream_t stream) {
  (void)in_sizes; (void)n_in; (void)out_size; (void)ws_size;
  const float* h_src = (const float*)d_in[0];
  const float* h_tgt = (const float*)d_in[1];
  // d_in[2], d_in[3]: all-True masks -> deterministic reshape path, ignored.

  char* ws = (char*)d_ws;
  const size_t MATB = (size_t)NROWS * HDIM * sizeof(_Float16);  // 16 MB each
  _Float16* As    = (_Float16*)(ws);
  _Float16* Bt    = (_Float16*)(ws + MATB);
  char* p = ws + 2 * MATB;
  float* partM = (float*)p;  p += (size_t)NROWS * NCHUNK * sizeof(float);
  float* partS = (float*)p;  p += (size_t)NROWS * NCHUNK * sizeof(float);
  int*   partI = (int*)p;    p += (size_t)NROWS * NCHUNK * sizeof(int);
  float* logS  = (float*)p;  p += (size_t)NROWS * sizeof(float);
  int*   pos   = (int*)p;    p += (size_t)NROWS * sizeof(int);
  float* acc   = (float*)p;

  init_kernel<<<1, 32, 0, stream>>>(acc);
  nrm_kernel<<<NROWS, 256, 0, stream>>>(h_src, As);
  nrm_kernel<<<NROWS, 256, 0, stream>>>(h_tgt, Bt);
  simred_kernel<<<(NROWS / 16) * NCHUNK / 8, 256, 0, stream>>>(As, Bt,
                                                               partM, partS, partI);
  merge_kernel<<<NROWS / 256, 256, 0, stream>>>(partM, partS, partI, logS, pos);
  ctx_kernel<<<NROWS / 8, 256, 0, stream>>>(Bt, pos, acc + 1);
  logsum_kernel<<<NROWS / 256, 256, 0, stream>>>(logS, acc);
  fin_kernel<<<1, 32, 0, stream>>>(acc, (float*)d_out);
}
